// TrajAR_88416196756301
// MI455X (gfx1250) — matrix operand order repaired and checked by reading, not compiled, not run on hardware
//
#include <hip/hip_runtime.h>

typedef __attribute__((ext_vector_type(8)))  float  v8f;
typedef __attribute__((ext_vector_type(8)))  __bf16 v8bf;
typedef __attribute__((ext_vector_type(16))) __bf16 v16bf;

#define TTOK  8192
#define DIN   512
#define HDIM  1024
#define ODIM  512
#define NEXP  16
#define NSH   3
#define NGRP  (NSH + NEXP)
#define TM    32        // tokens per block (two 16-row M tiles)

static __device__ __forceinline__ __bf16 f2bf(float f) {
  union { float f; unsigned u; } v; v.f = f;
  unsigned r = v.u + 0x7FFFu + ((v.u >> 16) & 1u);
  unsigned short h = (unsigned short)(r >> 16);
  return __builtin_bit_cast(__bf16, h);
}

static __device__ __forceinline__ v16bf cat16(v8bf lo, v8bf hi) {
  v16bf a;
  #pragma unroll
  for (int q = 0; q < 8; ++q) { a[q] = lo[q]; a[q + 8] = hi[q]; }
  return a;
}

// ---------------- fp32 -> bf16 row-major convert ----------------
__global__ void cvt_bf16_kernel(const float* __restrict__ src,
                                unsigned short* __restrict__ dst, long n) {
  long i = (long)blockIdx.x * blockDim.x + threadIdx.x;
  long stride = (long)gridDim.x * blockDim.x;
  for (; i < n; i += stride) {
    dst[i] = __builtin_bit_cast(unsigned short, f2bf(src[i]));
  }
}

// ------- pack fp32 [m][K][N] to WMMA-B bf16 layout [m][K/32][N/16][32][16] -------
// lane 0-15 : column n = lane,    K = kt*32 + 0..7, 16..23
// lane 16-31: column n = lane-16, K = kt*32 + 8..15, 24..31
__global__ void pack_b_kernel(const float* __restrict__ W,
                              unsigned short* __restrict__ P,
                              int Ktot, int Ntot, int nmat) {
  long total = (long)nmat * Ktot * Ntot;
  int NT = Ntot >> 4, KT = Ktot >> 5;
  long i = (long)blockIdx.x * blockDim.x + threadIdx.x;
  long stride = (long)gridDim.x * blockDim.x;
  for (; i < total; i += stride) {
    int  j    = (int)(i & 15);
    int  lane = (int)((i >> 4) & 31);
    long rest = i >> 9;
    int nt = (int)(rest % NT); rest /= NT;
    int kt = (int)(rest % KT);
    int m  = (int)(rest / KT);
    int k = kt * 32 + ((lane >> 4) << 3) + (j < 8 ? j : j + 8);
    int n = nt * 16 + (lane & 15);
    float v = W[(long)m * Ktot * Ntot + (long)k * Ntot + n];
    P[i] = __builtin_bit_cast(unsigned short, f2bf(v));
  }
}

// ---------------- gating: softmax -> top-2 -> renorm (x0.5 folded) ----------------
__global__ __launch_bounds__(256) void gate_kernel(
    const float* __restrict__ x, const float* __restrict__ gw,
    const float* __restrict__ gb, float* __restrict__ comb) {
  int lane = threadIdx.x & 31;
  int wv   = threadIdx.x >> 5;
  int t = blockIdx.x * 8 + wv;
  int e = lane & 15;
  const float* xr = x + (long)t * DIN;
  int d0 = (lane >> 4) * (DIN / 2);
  float part = 0.f;
  for (int d = 0; d < DIN / 2; ++d)
    part += xr[d0 + d] * gw[(long)(d0 + d) * NEXP + e];
  part += __shfl_xor(part, 16);            // both halves now hold full dot
  float logit = part + gb[e];
  float mx = logit;
  #pragma unroll
  for (int m = 8; m >= 1; m >>= 1) mx = fmaxf(mx, __shfl_xor(mx, m));
  float ex = __expf(logit - mx);
  float sm = ex;
  #pragma unroll
  for (int m = 8; m >= 1; m >>= 1) sm += __shfl_xor(sm, m);
  float p = ex / sm;
  // top-1 (ties -> lower index, matching lax.top_k)
  float v1 = p; int i1 = e;
  #pragma unroll
  for (int m = 8; m >= 1; m >>= 1) {
    float ov = __shfl_xor(v1, m); int oi = __shfl_xor(i1, m);
    if (ov > v1 || (ov == v1 && oi < i1)) { v1 = ov; i1 = oi; }
  }
  // top-2
  float pm = (e == i1) ? -1.f : p;
  float v2 = pm; int i2 = e;
  #pragma unroll
  for (int m = 8; m >= 1; m >>= 1) {
    float ov = __shfl_xor(v2, m); int oi = __shfl_xor(i2, m);
    if (ov > v2 || (ov == v2 && oi < i2)) { v2 = ov; i2 = oi; }
  }
  float wsum = v1 + v2;
  float wm = (e == i1) ? v1 / wsum : (e == i2) ? v2 / wsum : 0.f;
  if (lane < 16) comb[(long)t * NEXP + e] = 0.5f * wm;
}

// ---------------- fused expert FFN over 32-token tiles ----------------
__global__ __launch_bounds__(256) void moe_ffn_kernel(
    const unsigned short* __restrict__ xb_,   // [T][D] bf16
    const float* __restrict__ obj,            // [T][S]
    const float* __restrict__ comb,           // [T][E] (0.5-folded)
    const unsigned short* __restrict__ shw1p, const float* __restrict__ shb1,
    const unsigned short* __restrict__ shw2p, const float* __restrict__ shb2,
    const unsigned short* __restrict__ iw1p,  const float* __restrict__ ib1,
    const unsigned short* __restrict__ iw2p,  const float* __restrict__ ib2,
    float* __restrict__ out) {
  __shared__ __align__(16) __bf16 xs[TM][DIN + 8];   // ~33 KB, pad -> conflict-free
  __shared__ __align__(16) __bf16 hs[TM][128 + 8];   // ~8.7 KB
  const __bf16* xb = (const __bf16*)xb_;
  const int tid  = threadIdx.x;
  const int lane = tid & 31;
  const int wv   = tid >> 5;
  const int t0   = blockIdx.x * TM;

  // stage x tile: 32 rows x 512 bf16
  for (int i = tid; i < TM * (DIN / 8); i += 256) {
    int row = i >> 6;                       // DIN/8 == 64 8-elem chunks per row
    int c8  = i & 63;
    v8bf v = *(const v8bf*)(xb + (long)(t0 + row) * DIN + c8 * 8);
    *(v8bf*)(&xs[row][c8 * 8]) = v;
  }
  __syncthreads();

  const int n16    = lane & 15;
  const int rowoff = (lane >> 4) * 8;   // C-layout: lanes 16-31 hold M+8
  const int koff   = (lane >> 4) * 8;   // A-layout K sub-offset

  for (int g = 0; g < NGRP; ++g) {
    const __bf16 *w1b, *w2b;
    const float *b1, *b2;
    if (g < NSH) {
      w1b = (const __bf16*)shw1p + (long)g * DIN * HDIM;
      w2b = (const __bf16*)shw2p + (long)g * HDIM * ODIM;
      b1 = shb1 + (long)g * HDIM;
      b2 = shb2 + (long)g * ODIM;
    } else {
      int e = g - NSH;
      w1b = (const __bf16*)iw1p + (long)e * DIN * HDIM;
      w2b = (const __bf16*)iw2p + (long)e * HDIM * ODIM;
      b1 = ib1 + (long)e * HDIM;
      b2 = ib2 + (long)e * ODIM;
      float cs = comb[(long)(t0 + lane) * NEXP + e];   // 32 lanes -> 32 tokens
      if (!__any(cs > 0.f)) continue;                  // expert untouched
    }
    v8f acc[2][4];
    #pragma unroll
    for (int mi = 0; mi < 2; ++mi)
      #pragma unroll
      for (int nt = 0; nt < 4; ++nt) acc[mi][nt] = {};

    for (int hc = 0; hc < HDIM / 128; ++hc) {
      // ---- GEMM1: wave wv computes h columns [hc*128 + wv*16, +16) for both M tiles
      v8f c1[2] = {{}, {}};
      const int nt1 = hc * 8 + wv;         // NT1 = HDIM/16 = 64
      for (int kt = 0; kt < DIN / 32; ++kt) {
        const __bf16* bp = w1b + (((long)kt * 64 + nt1) * 32 + lane) * 16;
        v16bf b = cat16(*(const v8bf*)bp, *(const v8bf*)(bp + 8));
        #pragma unroll
        for (int mi = 0; mi < 2; ++mi) {
          const __bf16* xrow = &xs[mi * 16 + n16][0];
          v16bf a = cat16(*(const v8bf*)(xrow + kt * 32 + koff),
                          *(const v8bf*)(xrow + kt * 32 + koff + 16));
          c1[mi] = __builtin_amdgcn_wmma_f32_16x16x32_bf16(
              false, a, false, b, (short)0, c1[mi], false, false);
        }
      }
      float b1v = b1[hc * 128 + wv * 16 + n16];
      #pragma unroll
      for (int mi = 0; mi < 2; ++mi)
        #pragma unroll
        for (int r = 0; r < 8; ++r) {
          float hv = c1[mi][r] + b1v;
          hv = hv > 0.f ? hv : 0.f;
          hs[mi * 16 + r + rowoff][wv * 16 + n16] = f2bf(hv);
        }
      __syncthreads();
      // ---- GEMM2: accumulate this 128-wide chunk into the wave's 64 out cols ----
      #pragma unroll
      for (int ks = 0; ks < 4; ++ks) {
        v16bf a[2];
        #pragma unroll
        for (int mi = 0; mi < 2; ++mi) {
          const __bf16* hrow = &hs[mi * 16 + n16][0];
          a[mi] = cat16(*(const v8bf*)(hrow + ks * 32 + koff),
                        *(const v8bf*)(hrow + ks * 32 + koff + 16));
        }
        int kt2 = hc * 4 + ks;              // K index over HDIM/32 = 32
        #pragma unroll
        for (int nt = 0; nt < 4; ++nt) {
          const __bf16* bp = w2b + (((long)kt2 * 32 + (wv * 4 + nt)) * 32 + lane) * 16;
          v16bf b = cat16(*(const v8bf*)bp, *(const v8bf*)(bp + 8));
          #pragma unroll
          for (int mi = 0; mi < 2; ++mi)
            acc[mi][nt] = __builtin_amdgcn_wmma_f32_16x16x32_bf16(
                false, a[mi], false, b, (short)0, acc[mi][nt], false, false);
        }
      }
      __syncthreads();
    }
    // ---- epilogue: bias + per-token scale, write/accumulate output tile ----
    float scale[2][8];
    #pragma unroll
    for (int mi = 0; mi < 2; ++mi)
      #pragma unroll
      for (int r = 0; r < 8; ++r) {
        int t = t0 + mi * 16 + r + rowoff;
        scale[mi][r] = (g < NSH) ? 0.5f * obj[(long)t * NSH + g]
                                 : comb[(long)t * NEXP + (g - NSH)];
      }
    #pragma unroll
    for (int nt = 0; nt < 4; ++nt) {
      int ocol = wv * 64 + nt * 16 + n16;
      float b2v = b2[ocol];
      #pragma unroll
      for (int mi = 0; mi < 2; ++mi)
        #pragma unroll
        for (int r = 0; r < 8; ++r) {
          long oi = (long)(t0 + mi * 16 + r + rowoff) * ODIM + ocol;
          float v = scale[mi][r] * (acc[mi][nt][r] + b2v);
          if (g == 0) out[oi] = v; else out[oi] += v;
        }
    }
  }
}

extern "C" void kernel_launch(void* const* d_in, const int* in_sizes, int n_in,
                              void* d_out, int out_size, void* d_ws, size_t ws_size,
                              hipStream_t stream) {
  const float* x   = (const float*)d_in[0];
  const float* obj = (const float*)d_in[1];
  const float* gw  = (const float*)d_in[2];
  const float* gb  = (const float*)d_in[3];
  const float* sw1 = (const float*)d_in[4];
  const float* sb1 = (const float*)d_in[5];
  const float* sw2 = (const float*)d_in[6];
  const float* sb2 = (const float*)d_in[7];
  const float* iw1 = (const float*)d_in[8];
  const float* ib1 = (const float*)d_in[9];
  const float* iw2 = (const float*)d_in[10];
  const float* ib2 = (const float*)d_in[11];
  float* out = (float*)d_out;

  char* ws = (char*)d_ws;
  size_t off = 0;
  unsigned short* xb   = (unsigned short*)(ws + off); off += (size_t)TTOK * DIN * 2;
  float*          comb = (float*)(ws + off);          off += (size_t)TTOK * NEXP * 4;
  unsigned short* sw1p = (unsigned short*)(ws + off); off += (size_t)NSH * DIN * HDIM * 2;
  unsigned short* sw2p = (unsigned short*)(ws + off); off += (size_t)NSH * HDIM * ODIM * 2;
  unsigned short* iw1p = (unsigned short*)(ws + off); off += (size_t)NEXP * DIN * HDIM * 2;
  unsigned short* iw2p = (unsigned short*)(ws + off); off += (size_t)NEXP * HDIM * ODIM * 2;
  (void)ws_size; (void)in_sizes; (void)n_in; (void)out_size;

  cvt_bf16_kernel<<<dim3(2048), dim3(256), 0, stream>>>(x, xb, (long)TTOK * DIN);
  pack_b_kernel<<<dim3(2048), dim3(256), 0, stream>>>(sw1, sw1p, DIN, HDIM, NSH);
  pack_b_kernel<<<dim3(2048), dim3(256), 0, stream>>>(sw2, sw2p, HDIM, ODIM, NSH);
  pack_b_kernel<<<dim3(4096), dim3(256), 0, stream>>>(iw1, iw1p, DIN, HDIM, NEXP);
  pack_b_kernel<<<dim3(4096), dim3(256), 0, stream>>>(iw2, iw2p, HDIM, ODIM, NEXP);
  gate_kernel<<<dim3(TTOK / 8), dim3(256), 0, stream>>>(x, gw, gb, comb);
  moe_ffn_kernel<<<dim3(TTOK / TM), dim3(256), 0, stream>>>(
      xb, obj, comb, sw1p, sb1, sw2p, sb2, iw1p, ib1, iw2p, ib2, out);
}
